// FullModel_79585743995423
// MI455X (gfx1250) — matrix operand order repaired and checked by reading, not compiled, not run on hardware
//
#include <hip/hip_runtime.h>
#include <hip/hip_bf16.h>

typedef __attribute__((ext_vector_type(16))) _Float16 v16h;
typedef __attribute__((ext_vector_type(8)))  _Float16 v8h;
typedef __attribute__((ext_vector_type(2)))  _Float16 v2h;
typedef __attribute__((ext_vector_type(8)))  float    v8f;

// ---------------------------------------------------------------------------
// Generic f16-WMMA GEMM:  C[M,N] = act(A[M,K] * B + bias) (+ residual)
//   A indexed as A[row*lda + k]
//   B indexed as B[k*ldb_k + col*ldb_n]   (ldb_k=N,ldb_n=1 normal; 1,K transposed)
// Block: 256 threads = 8 waves; block tile 16(M) x 256(N); K step 32.
// Each wave owns two 16x16 output tiles (one A fragment, two WMMAs per step).
// B is staged TRANSPOSED in LDS so fragments are contiguous -> ds_load_b128.
// Compile-time ACT / HAS_BIAS / HAS_RES + a uniform full-tile fast path keep
// the epilogue branch-free for the dominant (M=12000, N%256==0) GEMMs.
// ---------------------------------------------------------------------------
template <int ACT, bool HAS_BIAS, bool HAS_RES>
__global__ __launch_bounds__(256) void gemm_f16wmma(
    const float* __restrict__ A, int lda,
    const float* __restrict__ B, int ldb_k, int ldb_n,
    const float* __restrict__ bias,
    const float* __restrict__ residual,
    float* __restrict__ C, int ldc,
    int M, int N, int K)
{
    __shared__ __align__(16) _Float16 As[16][40];    // [row][k], 16x32 (+pad)
    __shared__ __align__(16) _Float16 BsT[256][40];  // [col][k], 256x32 (+pad)

    const int tid  = threadIdx.x;
    const int lane = tid & 31;
    const int wave = tid >> 5;
    const int row0 = blockIdx.y * 16;
    const int col0 = blockIdx.x * 256;

    v8f acc0 = {}, acc1 = {};

    // A fragment: lanes 0-15 -> M=lane, k-halves [0..7]+[16..23];
    //             lanes 16-31 -> M=lane-16, k-halves [8..15]+[24..31]
    const int arow  = lane & 15;
    const int akoff = (lane < 16) ? 0 : 8;
    // B fragment: lane's column, k-halves [0..7]+[16..23] or [8..15]+[24..31]
    const int bcol0 = (wave << 5) + (lane & 15);
    const int bcol1 = bcol0 + 16;
    const int bkoff = (lane < 16) ? 0 : 8;

    for (int k0 = 0; k0 < K; k0 += 32) {
        // ---- stage A (16x32): one f32 pair per thread, packed b32 LDS store
        {
            int r  = tid >> 4;           // 0..15
            int c  = (tid & 15) << 1;    // 0,2,..,30
            int gr = row0 + r;
            int grc = (gr < M) ? gr : (M - 1);
            const float* ap = A + (long)grc * lda + (k0 + c);
            float f0 = ap[0];
            float f1 = ap[1];
            if (gr >= M) { f0 = 0.0f; f1 = 0.0f; }
            v2h p; p[0] = (_Float16)f0; p[1] = (_Float16)f1;
            *(v2h*)&As[r][c] = p;
        }
        // ---- stage B (32x256) transposed: 16 k-pairs per thread
        #pragma unroll
        for (int i = 0; i < 16; ++i) {
            int t  = i * 256 + tid;      // pair index 0..4095
            int rr = t >> 8;             // k-pair 0..15
            int c  = t & 255;            // col within tile
            int gc = col0 + c;
            int gcc = (gc < N) ? gc : (N - 1);
            long boff = (long)gcc * ldb_n;
            float f0 = B[(long)(k0 + 2 * rr)     * ldb_k + boff];
            float f1 = B[(long)(k0 + 2 * rr + 1) * ldb_k + boff];
            if (gc >= N) { f0 = 0.0f; f1 = 0.0f; }
            v2h p; p[0] = (_Float16)f0; p[1] = (_Float16)f1;
            *(v2h*)&BsT[c][2 * rr] = p;
        }
        __syncthreads();

        // ---- fragments: contiguous 16B chunks -> ds_load_b128
        v8h alo = *(const v8h*)&As[arow][akoff];
        v8h ahi = *(const v8h*)&As[arow][16 + akoff];
        v16h a = __builtin_shufflevector(alo, ahi,
                     0,1,2,3,4,5,6,7,8,9,10,11,12,13,14,15);

        v8h b0lo = *(const v8h*)&BsT[bcol0][bkoff];
        v8h b0hi = *(const v8h*)&BsT[bcol0][bkoff + 16];
        v16h b0 = __builtin_shufflevector(b0lo, b0hi,
                     0,1,2,3,4,5,6,7,8,9,10,11,12,13,14,15);

        v8h b1lo = *(const v8h*)&BsT[bcol1][bkoff];
        v8h b1hi = *(const v8h*)&BsT[bcol1][bkoff + 16];
        v16h b1 = __builtin_shufflevector(b1lo, b1hi,
                     0,1,2,3,4,5,6,7,8,9,10,11,12,13,14,15);

        acc0 = __builtin_amdgcn_wmma_f32_16x16x32_f16(
                   false, a, false, b0, (short)0, acc0, false, false);
        acc1 = __builtin_amdgcn_wmma_f32_16x16x32_f16(
                   false, a, false, b1, (short)0, acc1, false, false);
        __syncthreads();
    }

    // ---- epilogue: reg r -> row = r (+8 for upper half-wave); col = lane&15
    const int rbase = row0 + ((lane < 16) ? 0 : 8);
    const bool full = (row0 + 16 <= M) && (col0 + 256 <= N);  // uniform branch
    #pragma unroll
    for (int t2 = 0; t2 < 2; ++t2) {
        v8f acc = t2 ? acc1 : acc0;
        int ccol = col0 + (wave << 5) + (lane & 15) + t2 * 16;
        if (full) {
            // fast path: no per-element guards (all transformer GEMMs land here)
            float bv = HAS_BIAS ? bias[ccol] : 0.0f;
            #pragma unroll
            for (int r = 0; r < 8; ++r) {
                int grow = rbase + r;
                float v = acc[r] + bv;
                if (ACT == 1) { // tanh-approx GELU (jax.nn.gelu default)
                    float x3 = v * v * v;
                    v = 0.5f * v * (1.0f + tanhf(0.7978845608028654f *
                                                 (v + 0.044715f * x3)));
                }
                if (HAS_RES) v += residual[(long)grow * ldc + ccol];
                C[(long)grow * ldc + ccol] = v;
            }
        } else if (ccol < N) {
            float bv = HAS_BIAS ? bias[ccol] : 0.0f;
            #pragma unroll
            for (int r = 0; r < 8; ++r) {
                int grow = rbase + r;
                if (grow < M) {
                    float v = acc[r] + bv;
                    if (ACT == 1) {
                        float x3 = v * v * v;
                        v = 0.5f * v * (1.0f + tanhf(0.7978845608028654f *
                                                     (v + 0.044715f * x3)));
                    }
                    if (HAS_RES) v += residual[(long)grow * ldc + ccol];
                    C[(long)grow * ldc + ccol] = v;
                }
            }
        }
    }
}

// ---------------------------------------------------------------------------
// LayerNorm over rows of 512; one wave per row (wave32), 8 rows per block.
// ---------------------------------------------------------------------------
__global__ __launch_bounds__(256) void layernorm_rows(
    const float* __restrict__ X, const float* __restrict__ g,
    const float* __restrict__ bta, float* __restrict__ Y, int rows)
{
    int lane = threadIdx.x & 31;
    int wave = threadIdx.x >> 5;
    int row  = blockIdx.x * 8 + wave;
    if (row >= rows) return;
    const float* xr = X + (long)row * 512;
    float v[16], s = 0.0f, sq = 0.0f;
    #pragma unroll
    for (int j = 0; j < 16; ++j) {
        v[j] = xr[j * 32 + lane];
        s += v[j]; sq += v[j] * v[j];
    }
    #pragma unroll
    for (int o = 16; o > 0; o >>= 1) {
        s  += __shfl_xor(s,  o, 32);
        sq += __shfl_xor(sq, o, 32);
    }
    float m    = s  * (1.0f / 512.0f);
    float var  = sq * (1.0f / 512.0f) - m * m;
    float rinv = rsqrtf(var + 1e-5f);
    float* yr = Y + (long)row * 512;
    #pragma unroll
    for (int j = 0; j < 16; ++j) {
        int c = j * 32 + lane;
        yr[c] = (v[j] - m) * rinv * g[c] + bta[c];
    }
}

// ---------------------------------------------------------------------------
// Causal MHA for L=12, DH=64; one wave per (n,h) pair; 8 waves per block.
// qkv: [1000,12,1536] (q|k|v), out: [1000,12,512], attn: [1000,8,12,12]
// ---------------------------------------------------------------------------
__global__ __launch_bounds__(256) void attention_kernel(
    const float* __restrict__ qkv,
    float* __restrict__ out,
    float* __restrict__ attn)
{
    __shared__ float sS[8][12][13];
    int lane = threadIdx.x & 31;
    int wave = threadIdx.x >> 5;
    int pair = blockIdx.x * 8 + wave;   // 0..7999
    int n = pair >> 3;
    int h = pair & 7;
    const float* base = qkv + (long)n * 12 * 1536 + h * 64;

    // scores (144 entries spread over 32 lanes)
    for (int t = lane; t < 144; t += 32) {
        int i = t / 12, j = t % 12;
        const float* q = base + (long)i * 1536;
        const float* k = base + (long)j * 1536 + 512;
        float d = 0.0f;
        #pragma unroll 16
        for (int c = 0; c < 64; ++c) d += q[c] * k[c];
        d *= 0.125f;                       // 1/sqrt(64)
        if (j > i) d -= 1e9f;              // causal mask
        sS[wave][i][j] = d;
    }
    __syncthreads();

    // softmax per query row (lanes 0..11)
    if (lane < 12) {
        int i = lane;
        float r[12], mx = -3.4e38f;
        #pragma unroll
        for (int j = 0; j < 12; ++j) { r[j] = sS[wave][i][j]; mx = fmaxf(mx, r[j]); }
        float sum = 0.0f;
        #pragma unroll
        for (int j = 0; j < 12; ++j) { r[j] = expf(r[j] - mx); sum += r[j]; }
        float inv = 1.0f / sum;
        #pragma unroll
        for (int j = 0; j < 12; ++j) {
            float p = r[j] * inv;
            sS[wave][i][j] = p;
            attn[(((long)n * 8 + h) * 12 + i) * 12 + j] = p;
        }
    }
    __syncthreads();

    // out = attn @ V  (768 = 12x64 outputs per wave)
    for (int t = lane; t < 768; t += 32) {
        int i = t >> 6, d = t & 63;
        const float* v = base + 1024 + d;
        float o = 0.0f;
        #pragma unroll
        for (int j = 0; j < 12; ++j) o += sS[wave][i][j] * v[(long)j * 1536];
        out[((long)n * 12 + i) * 512 + h * 64 + d] = o;
    }
}

// ---------------------------------------------------------------------------
// Row L2 normalize (cols=512); optional scale by exp(*logit_scale) for images.
// ---------------------------------------------------------------------------
__global__ __launch_bounds__(256) void rownorm_kernel(
    const float* __restrict__ X, float* __restrict__ Y, int rows,
    const float* __restrict__ logit_scale)
{
    int lane = threadIdx.x & 31;
    int wave = threadIdx.x >> 5;
    int row  = blockIdx.x * 8 + wave;
    if (row >= rows) return;
    const float* xr = X + (long)row * 512;
    float v[16], sq = 0.0f;
    #pragma unroll
    for (int j = 0; j < 16; ++j) { v[j] = xr[j * 32 + lane]; sq += v[j] * v[j]; }
    #pragma unroll
    for (int o = 16; o > 0; o >>= 1) sq += __shfl_xor(sq, o, 32);
    float inv = rsqrtf(sq);
    if (logit_scale) inv *= expf(logit_scale[0]);
    float* yr = Y + (long)row * 512;
    #pragma unroll
    for (int j = 0; j < 16; ++j) yr[j * 32 + lane] = v[j] * inv;
}

// attr[n,p] = mean_h attn[n,h,11,p], p<5; normalized over p (+1e-8)
__global__ __launch_bounds__(32) void attr_kernel(
    const float* __restrict__ attn, float* __restrict__ attr)
{
    int n = blockIdx.x, lane = threadIdx.x;
    float a = 0.0f;
    if (lane < 5) {
        for (int h = 0; h < 8; ++h)
            a += attn[(((long)n * 8 + h) * 12 + 11) * 12 + lane];
        a *= 0.125f;
    }
    float t = (lane < 5) ? a : 0.0f;
    #pragma unroll
    for (int o = 16; o > 0; o >>= 1) t += __shfl_xor(t, o, 32);
    if (lane < 5) attr[n * 5 + lane] = a / (t + 1e-8f);
}

// adjusted prompt: x[n,t,:] = raw[n,t,:] * (t<5 ? attr[n,t] : 1)
__global__ __launch_bounds__(256) void adjust_kernel(
    const float* __restrict__ raw, const float* __restrict__ attr,
    float* __restrict__ x)
{
    int n = blockIdx.x;
    for (int e = threadIdx.x; e < 6144; e += 256) {
        int t = e >> 9;
        float s = (t < 5) ? attr[n * 5 + t] : 1.0f;
        x[(long)n * 6144 + e] = raw[(long)n * 6144 + e] * s;
    }
}

// attribution output: broadcast mean_n attr[n,p] to [256,5]
__global__ __launch_bounds__(256) void attribution_out_kernel(
    const float* __restrict__ attr, float* __restrict__ out)
{
    __shared__ float m[5];
    int tid = threadIdx.x;
    if (tid < 5) {
        float s = 0.0f;
        for (int n = 0; n < 1000; ++n) s += attr[n * 5 + tid];
        m[tid] = s * 1e-3f;
    }
    __syncthreads();
    #pragma unroll
    for (int p = 0; p < 5; ++p) out[tid * 5 + p] = m[p];
}

// ---------------------------------------------------------------------------
extern "C" void kernel_launch(void* const* d_in, const int* in_sizes, int n_in,
                              void* d_out, int out_size, void* d_ws, size_t ws_size,
                              hipStream_t stream)
{
    (void)in_sizes; (void)n_in; (void)out_size; (void)ws_size;
    const float* images     = (const float*)d_in[0];
    const float* raw_prompt = (const float*)d_in[1];
    const float* Wqkv       = (const float*)d_in[2];
    const float* bqkv       = (const float*)d_in[3];
    const float* Wo         = (const float*)d_in[4];
    const float* bo         = (const float*)d_in[5];
    const float* W1         = (const float*)d_in[6];
    const float* b1         = (const float*)d_in[7];
    const float* W2         = (const float*)d_in[8];
    const float* b2         = (const float*)d_in[9];
    const float* ln1_g      = (const float*)d_in[10];
    const float* ln1_b      = (const float*)d_in[11];
    const float* ln2_g      = (const float*)d_in[12];
    const float* ln2_b      = (const float*)d_in[13];
    const float* Wp         = (const float*)d_in[14];
    const float* logit_scale= (const float*)d_in[15];
    float* out = (float*)d_out;

    // workspace layout (floats)
    float* ws   = (float*)d_ws;
    size_t o = 0;
    float* x    = ws + o; o += 6144000;   // [1000,12,512]
    float* xln  = ws + o; o += 6144000;   // LN output / attention output
    float* big  = ws + o; o += 24576000;  // qkv [.,1536] then FFN hidden [.,2048]
    float* attn = ws + o; o += 1152000;   // [1000,8,12,12]
    float* attr = ws + o; o += 5000;      // [1000,5]
    float* tf   = ws + o; o += 512000;    // text features [1000,512]
    float* img  = ws + o; o += 131072;    // scaled image features [256,512]

    const int M = 12000;  // 1000 sequences * 12 tokens

    // bias + no-res (ACT=0): QKV
    auto gemm_b = [&](const float* A, int lda, const float* B, int ldbk, int ldbn,
                      const float* bias, float* C, int ldc, int Mm, int Nn, int Kk) {
        dim3 g((Nn + 255) / 256, (Mm + 15) / 16);
        gemm_f16wmma<0, true, false><<<g, 256, 0, stream>>>(
            A, lda, B, ldbk, ldbn, bias, nullptr, C, ldc, Mm, Nn, Kk);
    };
    // bias + residual (ACT=0): out-proj, FFN2
    auto gemm_br = [&](const float* A, int lda, const float* B, int ldbk, int ldbn,
                       const float* bias, const float* res, float* C, int ldc,
                       int Mm, int Nn, int Kk) {
        dim3 g((Nn + 255) / 256, (Mm + 15) / 16);
        gemm_f16wmma<0, true, true><<<g, 256, 0, stream>>>(
            A, lda, B, ldbk, ldbn, bias, res, C, ldc, Mm, Nn, Kk);
    };
    // bias + GELU: FFN1
    auto gemm_bg = [&](const float* A, int lda, const float* B, int ldbk, int ldbn,
                       const float* bias, float* C, int ldc, int Mm, int Nn, int Kk) {
        dim3 g((Nn + 255) / 256, (Mm + 15) / 16);
        gemm_f16wmma<1, true, false><<<g, 256, 0, stream>>>(
            A, lda, B, ldbk, ldbn, bias, nullptr, C, ldc, Mm, Nn, Kk);
    };
    // plain: text projection, logits
    auto gemm_p = [&](const float* A, int lda, const float* B, int ldbk, int ldbn,
                      float* C, int ldc, int Mm, int Nn, int Kk) {
        dim3 g((Nn + 255) / 256, (Mm + 15) / 16);
        gemm_f16wmma<0, false, false><<<g, 256, 0, stream>>>(
            A, lda, B, ldbk, ldbn, nullptr, nullptr, C, ldc, Mm, Nn, Kk);
    };

    auto transformer = [&]() {
        for (int l = 0; l < 2; ++l) {
            layernorm_rows<<<1500, 256, 0, stream>>>(x, ln1_g + l*512, ln1_b + l*512, xln, M);
            gemm_b(xln, 512, Wqkv + (long)l*512*1536, 1536, 1,
                   bqkv + l*1536, big, 1536, M, 1536, 512);
            attention_kernel<<<1000, 256, 0, stream>>>(big, xln, attn);
            gemm_br(xln, 512, Wo + (long)l*512*512, 512, 1,
                    bo + l*512, x, x, 512, M, 512, 512);
            layernorm_rows<<<1500, 256, 0, stream>>>(x, ln2_g + l*512, ln2_b + l*512, xln, M);
            gemm_bg(xln, 512, W1 + (long)l*512*2048, 2048, 1,
                    b1 + l*2048, big, 2048, M, 2048, 512);
            gemm_br(big, 2048, W2 + (long)l*2048*512, 512, 1,
                    b2 + l*512, x, x, 512, M, 512, 2048);
        }
    };

    // ---- pass 1 on raw prompt ----
    hipMemcpyAsync((void*)x, (const void*)raw_prompt, sizeof(float) * 6144000,
                   hipMemcpyDeviceToDevice, stream);
    transformer();

    // ---- attribution from last-layer attention ----
    attr_kernel<<<1000, 32, 0, stream>>>(attn, attr);
    adjust_kernel<<<1000, 256, 0, stream>>>(raw_prompt, attr, x);

    // ---- pass 2 on adjusted prompt ----
    transformer();

    // ---- text features: x2[:,11,:] @ Wp, then L2-normalize ----
    gemm_p(x + 11 * 512, 6144, Wp, 512, 1, tf, 512, 1000, 512, 512);
    rownorm_kernel<<<(1000 + 7) / 8, 256, 0, stream>>>(tf, tf, 1000, nullptr);

    // ---- image features scaled by exp(logit_scale) ----
    rownorm_kernel<<<(256 + 7) / 8, 256, 0, stream>>>(images, img, 256, logit_scale);

    // ---- logits = img_scaled @ tf^T  (B transposed: ldb_k=1, ldb_n=512) ----
    gemm_p(img, 512, tf, 1, 512, out, 1000, 256, 1000, 512);

    // ---- attribution broadcast [256,5] ----
    attribution_out_kernel<<<1, 256, 0, stream>>>(attr, out + 256000);
}
